// BlockwiseAttention_51891794870903
// MI455X (gfx1250) — compile-verified, matrix operand order checked by reading, not run on hardware
//
#include <hip/hip_runtime.h>
#include <hip/hip_bf16.h>
#include <cstdint>

// Problem constants (from reference)
#define N_B   4
#define S_L   2048
#define E_D   768
#define H_N   12
#define HD_D  64
#define BS_B  128
#define NB_B  16
#define M_ROWS (N_B * S_L)          // 8192

typedef __attribute__((ext_vector_type(16))) __bf16          v16bf;
typedef __attribute__((ext_vector_type(16))) unsigned short  v16us;
typedef __attribute__((ext_vector_type(8)))  float           v8f;

// Packed (hi,lo) bf16 decomposition of an f32: hi = RNE bf16(v) in [15:0],
// lo = RNE bf16(v - hi) in [31:16].
static __device__ __forceinline__ unsigned pack_bf16x2(float v) {
    unsigned u  = __float_as_uint(v);
    unsigned hi = (u + 0x7FFFu + ((u >> 16) & 1u)) >> 16;
    float    hf = __uint_as_float(hi << 16);
    float    lo = v - hf;
    unsigned ul = __float_as_uint(lo);
    unsigned lb = (ul + 0x7FFFu + ((ul >> 16) & 1u)) >> 16;
    return (hi & 0xFFFFu) | (lb << 16);
}

// Fragment-ready position of K index k inside a row's LDS strip.
// Per CDNA5 16-bit A/B layout: lanes 0-15 hold K {0..7,16..23} of each 32-chunk,
// lanes 16-31 hold K {8..15,24..31}. Within a 32-chunk: [half*16 + i].
static __device__ __forceinline__ int fragpos(int k) {
    int kk = k & 31;
    return (k >> 5) * 32 + ((kk >> 3) & 1) * 16 + (kk & 7) + ((kk & 16) ? 8 : 0);
}

// 32-byte-aligned contiguous fragment load: lowers to 2x ds_load_b128, no shuffles.
static __device__ __forceinline__ v16bf load_frag(const unsigned short* base) {
    return __builtin_bit_cast(v16bf, *(const v16us*)base);
}

static __device__ __forceinline__ v8f v8f_zero() {
    v8f z = {0.f, 0.f, 0.f, 0.f, 0.f, 0.f, 0.f, 0.f};
    return z;
}

// Store a pair of packed elements (even pos) into hi/lo planes as single u32 stores.
static __device__ __forceinline__ void store_pair(unsigned short* hi_plane,
                                                  unsigned short* lo_plane,
                                                  int elem, unsigned u0, unsigned u1) {
    *(unsigned*)&hi_plane[elem] = (u0 & 0xFFFFu) | (u1 << 16);
    *(unsigned*)&lo_plane[elem] = (u0 >> 16)     | (u1 & 0xFFFF0000u);
}

// ---------------------------------------------------------------------------
// Y[m, j] = sum_e X[m, e] * W[j, e] + bias[j]
// 128x128 block tile, K chunks of 32, bf16 hi/lo split:
// 3 x v_wmma_f32_16x16x32_bf16 per (tile, chunk).
// ---------------------------------------------------------------------------
#define GSTRIDE 48   // row stride in u16 for 32-wide planes (96 B = 3*32, 16 u16 pad)

__global__ __launch_bounds__(256)
void gemm_bf16x3(const float* __restrict__ X, const float* __restrict__ W,
                 const float* __restrict__ bias, float* __restrict__ Y) {
    __shared__ alignas(64) unsigned short a_hi[128 * GSTRIDE];
    __shared__ alignas(64) unsigned short a_lo[128 * GSTRIDE];
    __shared__ alignas(64) unsigned short b_hi[128 * GSTRIDE];
    __shared__ alignas(64) unsigned short b_lo[128 * GSTRIDE];

    const int tid      = threadIdx.x;
    const int wave     = tid >> 5;
    const int lane     = tid & 31;
    const int laneHalf = lane >> 4;
    const int lmod     = lane & 15;
    const int m_blk    = blockIdx.y * 128;
    const int n_blk    = blockIdx.x * 128;

    v8f acc[8];
#pragma unroll
    for (int t = 0; t < 8; ++t) acc[t] = v8f_zero();

    for (int k0 = 0; k0 < E_D; k0 += 32) {
        // Stage this K chunk, pre-swizzled into fragment order; 2048 element-pairs.
#pragma unroll 2
        for (int idx = tid; idx < 2048; idx += 256) {
            const int r   = idx >> 4;
            const int k   = (idx & 15) * 2;
            const int pos = fragpos(k);
            {
                const float* px = &X[(size_t)(m_blk + r) * E_D + k0 + k];
                store_pair(a_hi, a_lo, r * GSTRIDE + pos,
                           pack_bf16x2(px[0]), pack_bf16x2(px[1]));
            }
            {
                const float* pw = &W[(size_t)(n_blk + r) * E_D + k0 + k];
                store_pair(b_hi, b_lo, r * GSTRIDE + pos,
                           pack_bf16x2(pw[0]), pack_bf16x2(pw[1]));
            }
        }
        __syncthreads();

        const int abase = (wave * 16 + lmod) * GSTRIDE + laneHalf * 16;
        v16bf ah = load_frag(&a_hi[abase]);
        v16bf al = load_frag(&a_lo[abase]);

#pragma unroll
        for (int t = 0; t < 8; ++t) {
            const int bbase = (t * 16 + lmod) * GSTRIDE + laneHalf * 16;
            v16bf bh = load_frag(&b_hi[bbase]);
            v16bf bl = load_frag(&b_lo[bbase]);
            acc[t] = __builtin_amdgcn_wmma_f32_16x16x32_bf16(false, ah, false, bh, (short)0, acc[t], false, false);
            acc[t] = __builtin_amdgcn_wmma_f32_16x16x32_bf16(false, ah, false, bl, (short)0, acc[t], false, false);
            acc[t] = __builtin_amdgcn_wmma_f32_16x16x32_bf16(false, al, false, bh, (short)0, acc[t], false, false);
        }
        __syncthreads();
    }

    // D layout: VGPR r -> M = 16*wave + laneHalf*8 + r ; N = tile*16 + lmod
#pragma unroll
    for (int t = 0; t < 8; ++t) {
        const int ncol = n_blk + t * 16 + lmod;
        const float bv = bias[ncol];
#pragma unroll
        for (int r = 0; r < 8; ++r) {
            const int mrow = m_blk + wave * 16 + laneHalf * 8 + r;
            Y[(size_t)mrow * E_D + ncol] = acc[t][r] + bv;
        }
    }
}

// ---------------------------------------------------------------------------
// One workgroup per (n, h, block): scores = Q Kt / 8 -> softmax -> diagonal
// block of w_full -> P @ V -> attn_out (row-major [N*S, E]).
// LDS arena (149,504 B) phased:
//   region A [0, 67584): f32 scores [128][132]  -> later V hi/lo planes
//   region B [67584, ..): Q/K hi/lo planes      -> later P hi/lo planes
// ---------------------------------------------------------------------------
#define QSTRIDE 80    // 64 data + 16 pad u16 (160 B = 5*32)
#define PSTRIDE 144   // 128 data + 16 pad u16 (288 B = 9*32)
#define SC_STRIDE 132 // f32 scores row stride
#define ARENA_A   67584      // 128*132*4
#define ARENA_SZ  149504     // ARENA_A + 4*128*QSTRIDE*2

__global__ __launch_bounds__(256)
void block_attention(const float* __restrict__ Qg, const float* __restrict__ Kg,
                     const float* __restrict__ Vg, float* __restrict__ wfull,
                     float* __restrict__ attn_out) {
    __shared__ alignas(64) unsigned char arena[ARENA_SZ];

    float*          sc   = reinterpret_cast<float*>(arena);
    unsigned short* v_hi = reinterpret_cast<unsigned short*>(arena);           // phase 4+
    unsigned short* v_lo = v_hi + 64 * PSTRIDE;
    unsigned short* q_hi = reinterpret_cast<unsigned short*>(arena + ARENA_A); // phase 1-2
    unsigned short* q_lo = q_hi + 128 * QSTRIDE;
    unsigned short* k_hi = q_lo + 128 * QSTRIDE;
    unsigned short* k_lo = k_hi + 128 * QSTRIDE;
    unsigned short* p_hi = reinterpret_cast<unsigned short*>(arena + ARENA_A); // phase 3+
    unsigned short* p_lo = p_hi + 128 * PSTRIDE;

    const int blk = blockIdx.x;
    const int b   = blk % NB_B;
    const int h   = (blk / NB_B) % H_N;
    const int n   = blk / (NB_B * H_N);

    const int tid      = threadIdx.x;
    const int wave     = tid >> 5;
    const int lane     = tid & 31;
    const int laneHalf = lane >> 4;
    const int lmod     = lane & 15;

    const size_t row0 = (size_t)n * S_L + (size_t)b * BS_B;
    const float* Qb = Qg + row0 * E_D + h * HD_D;   // [128 x 64], row stride E_D
    const float* Kb = Kg + row0 * E_D + h * HD_D;
    const float* Vb = Vg + row0 * E_D + h * HD_D;

    // ---- Phase 1: stage Q and K into fragment-ready hi/lo planes ----
#pragma unroll 2
    for (int idx = tid; idx < 8192; idx += 256) {   // 2 matrices x 128 rows x 32 pairs
        const int r   = (idx & 4095) >> 5;
        const int k   = (idx & 31) * 2;
        const int pos = fragpos(k);
        const bool isQ = idx < 4096;
        const float* src = (isQ ? Qb : Kb) + (size_t)r * E_D + k;
        unsigned short* hp = isQ ? q_hi : k_hi;
        unsigned short* lp = isQ ? q_lo : k_lo;
        store_pair(hp, lp, r * QSTRIDE + pos, pack_bf16x2(src[0]), pack_bf16x2(src[1]));
    }
    __syncthreads();

    // ---- Phase 2: scores = Q @ K^T (128x128, K=64) ----
    v8f acc[8];
#pragma unroll
    for (int t = 0; t < 8; ++t) acc[t] = v8f_zero();

#pragma unroll
    for (int c = 0; c < 2; ++c) {                   // K chunks of 32
        const int abase = (wave * 16 + lmod) * QSTRIDE + c * 32 + laneHalf * 16;
        v16bf ah = load_frag(&q_hi[abase]);
        v16bf al = load_frag(&q_lo[abase]);
#pragma unroll
        for (int t = 0; t < 8; ++t) {
            const int bbase = (t * 16 + lmod) * QSTRIDE + c * 32 + laneHalf * 16;
            v16bf bh = load_frag(&k_hi[bbase]);
            v16bf bl = load_frag(&k_lo[bbase]);
            acc[t] = __builtin_amdgcn_wmma_f32_16x16x32_bf16(false, ah, false, bh, (short)0, acc[t], false, false);
            acc[t] = __builtin_amdgcn_wmma_f32_16x16x32_bf16(false, ah, false, bl, (short)0, acc[t], false, false);
            acc[t] = __builtin_amdgcn_wmma_f32_16x16x32_bf16(false, al, false, bh, (short)0, acc[t], false, false);
        }
    }

    const float scale = 0.125f;   // 1/sqrt(64)
#pragma unroll
    for (int t = 0; t < 8; ++t) {
#pragma unroll
        for (int r = 0; r < 8; ++r) {
            const int qr = wave * 16 + laneHalf * 8 + r;
            sc[qr * SC_STRIDE + t * 16 + lmod] = acc[t][r] * scale;
        }
    }
    __syncthreads();

    // ---- Phase 3: softmax per row; emit w_full diag block; repack P planes ----
    if (tid < 128) {
        const float* srow = &sc[tid * SC_STRIDE];
        float mx = -3.402823466e+38f;
        for (int j = 0; j < 128; ++j) mx = fmaxf(mx, srow[j]);
        float sum = 0.f;
        for (int j = 0; j < 128; ++j) sum += __expf(srow[j] - mx);
        const float inv = 1.f / sum;
        float* wrow = wfull + (((size_t)(n * H_N + h) * S_L) + (size_t)b * BS_B + tid) * S_L
                            + (size_t)b * BS_B;
        for (int j = 0; j < 128; j += 2) {
            float p0 = __expf(srow[j]     - mx) * inv;
            float p1 = __expf(srow[j + 1] - mx) * inv;
            __builtin_nontemporal_store(p0, wrow + j);      // streaming 805MB output
            __builtin_nontemporal_store(p1, wrow + j + 1);
            store_pair(p_hi, p_lo, tid * PSTRIDE + fragpos(j),
                       pack_bf16x2(p0), pack_bf16x2(p1));
        }
    }
    __syncthreads();

    // ---- Phase 4: stage V (transposed) into fragment-ready planes over scores ----
#pragma unroll 2
    for (int idx = tid; idx < 4096; idx += 256) {   // 64 cols x 64 k-pairs
        const int d   = idx >> 6;
        const int k   = (idx & 63) * 2;
        const int pos = fragpos(k);
        const float x0 = Vb[(size_t)k       * E_D + d];
        const float x1 = Vb[(size_t)(k + 1) * E_D + d];
        store_pair(v_hi, v_lo, d * PSTRIDE + pos, pack_bf16x2(x0), pack_bf16x2(x1));
    }
    __syncthreads();

    // ---- Phase 5: out = P @ V (128x64, K=128) ----
    v8f oacc[4];
#pragma unroll
    for (int t = 0; t < 4; ++t) oacc[t] = v8f_zero();

#pragma unroll
    for (int c = 0; c < 4; ++c) {
        const int abase = (wave * 16 + lmod) * PSTRIDE + c * 32 + laneHalf * 16;
        v16bf ah = load_frag(&p_hi[abase]);
        v16bf al = load_frag(&p_lo[abase]);
#pragma unroll
        for (int t = 0; t < 4; ++t) {
            const int bbase = (t * 16 + lmod) * PSTRIDE + c * 32 + laneHalf * 16;
            v16bf bh = load_frag(&v_hi[bbase]);
            v16bf bl = load_frag(&v_lo[bbase]);
            oacc[t] = __builtin_amdgcn_wmma_f32_16x16x32_bf16(false, ah, false, bh, (short)0, oacc[t], false, false);
            oacc[t] = __builtin_amdgcn_wmma_f32_16x16x32_bf16(false, ah, false, bl, (short)0, oacc[t], false, false);
            oacc[t] = __builtin_amdgcn_wmma_f32_16x16x32_bf16(false, al, false, bh, (short)0, oacc[t], false, false);
        }
    }

#pragma unroll
    for (int t = 0; t < 4; ++t) {
#pragma unroll
        for (int r = 0; r < 8; ++r) {
            const int qr = wave * 16 + laneHalf * 8 + r;
            attn_out[(row0 + qr) * E_D + h * HD_D + t * 16 + lmod] = oacc[t][r];
        }
    }
}

extern "C" void kernel_launch(void* const* d_in, const int* in_sizes, int n_in,
                              void* d_out, int out_size, void* d_ws, size_t ws_size,
                              hipStream_t stream) {
    (void)in_sizes; (void)n_in; (void)out_size; (void)ws_size;
    const float* x  = (const float*)d_in[0];
    const float* Wq = (const float*)d_in[1];
    const float* bq = (const float*)d_in[2];
    const float* Wk = (const float*)d_in[3];
    const float* bk = (const float*)d_in[4];
    const float* Wv = (const float*)d_in[5];
    const float* bv = (const float*)d_in[6];
    const float* Wf = (const float*)d_in[7];
    const float* bf = (const float*)d_in[8];

    float* out   = (float*)d_out;                          // [N,S,E]
    const size_t nse = (size_t)N_B * S_L * E_D;            // 6,291,456
    float* wfull = out + nse;                              // [N,H,S,S]
    const size_t wfull_elems = (size_t)N_B * H_N * S_L * (size_t)S_L;

    float* Qg   = (float*)d_ws;                            // each [N*S, E] row-major
    float* Kg   = Qg + nse;
    float* Vg   = Kg + nse;
    float* attn = Vg + nse;                                // ~100.7 MB of d_ws

    // Zero the 805 MB block-diagonal weight output; diagonal blocks overwritten below.
    hipMemsetAsync(wfull, 0, wfull_elems * sizeof(float), stream);

    dim3 gg(E_D / 128, M_ROWS / 128);                      // (6, 64)
    gemm_bf16x3<<<gg, 256, 0, stream>>>(x, Wq, bq, Qg);
    gemm_bf16x3<<<gg, 256, 0, stream>>>(x, Wk, bk, Kg);
    gemm_bf16x3<<<gg, 256, 0, stream>>>(x, Wv, bv, Vg);

    block_attention<<<N_B * H_N * NB_B, 256, 0, stream>>>(Qg, Kg, Vg, wfull, attn);

    gemm_bf16x3<<<gg, 256, 0, stream>>>(attn, Wf, bf, out);
}
